// Conv2d_Local_54778012893893
// MI455X (gfx1250) — compile-verified, matrix operand order checked by reading, not run on hardware
//
#include <hip/hip_runtime.h>

// Locally-connected conv as 1521 independent GEMMs:
//   D_p[128x64] = A_p[128x784] * B_p[64x784]^T   (K = 28x28 kernel window)
// F16 WMMA 16x16x32 path (fp32 accumulate). Each kernel row (28 wide) is
// zero-padded to 32 in LDS so one K-step == one kernel row, giving aligned
// 16B fragment loads. Staging converts fp32 -> f16 with v_cvt_pk_rtz_f16_f32.

#define B_    128
#define H_    560
#define W_    560
#define KH_   28
#define KW_   28
#define DH_   14
#define DW_   14
#define NF_   64
#define OH_   39
#define OW_   39
#define P_    (OH_ * OW_)      // 1521
#define K_    (KH_ * KW_)      // 784
#define NFP_  (NF_ * P_)       // 97344

#define AROW  136              // 128 padded K-halves + 8 skew (bank-conflict free)
#define BROW  136
#define RPC   4                // kernel rows per LDS chunk
#define NCHUNK (KH_ / RPC)     // 7

typedef _Float16 v16h  __attribute__((ext_vector_type(16)));
typedef __fp16   v2fp16 __attribute__((ext_vector_type(2)));
typedef float    v8f   __attribute__((ext_vector_type(8)));

union Frag { uint4 q[2]; v16h v; };
union PackH2 { v2fp16 h2; unsigned u; };

__device__ __forceinline__ unsigned pkcvt(float a, float b) {
    PackH2 t; t.h2 = __builtin_amdgcn_cvt_pkrtz(a, b);  // v_cvt_pk_rtz_f16_f32
    return t.u;
}

__global__ __launch_bounds__(256) void lc_conv_wmma(
    const float* __restrict__ x, const float* __restrict__ w,
    const float* __restrict__ bias, float* __restrict__ out)
{
    __shared__ __align__(16) _Float16 As[B_  * AROW];  // 34816 B
    __shared__ __align__(16) _Float16 Bs[NF_ * BROW];  // 17408 B

    const int p    = blockIdx.x;
    const int oh   = p / OW_, ow = p % OW_;
    const int tid  = threadIdx.x;
    const int lane = tid & 31;
    const int wave = tid >> 5;       // wave -> batch rows [16*wave, 16*wave+16)
    const int nlo  = lane & 15;
    const int hi   = lane >> 4;      // upper half-wave: +8 half offset in K runs
    const int lbase = hi * 8;        // halves

    v8f acc0 = {}, acc1 = {}, acc2 = {}, acc3 = {};

    for (int chunk = 0; chunk < NCHUNK; ++chunk) {
        const int kh0 = chunk * RPC;

        // ---- stage A: 128 batches x 4 kernel rows of x, fp32 -> f16, pad 28->32
        // x rows are only 8B aligned (ow*14*4 = ow*56), so float2 loads.
        for (int l = tid; l < B_ * RPC; l += 256) {
            const int b   = l >> 2;
            const int khl = l & 3;
            const float* src = x + (size_t)b * (H_ * W_)
                                 + (size_t)(oh * DH_ + kh0 + khl) * W_
                                 + (size_t)(ow * DW_);
            _Float16* dst = &As[b * AROW + khl * 32];
            #pragma unroll
            for (int j = 0; j < 7; ++j) {
                float2 v0 = *(const float2*)(src + 4 * j);
                float2 v1 = *(const float2*)(src + 4 * j + 2);
                uint2 pkd; pkd.x = pkcvt(v0.x, v0.y); pkd.y = pkcvt(v1.x, v1.y);
                *(uint2*)(dst + 4 * j) = pkd;
            }
            *(uint2*)(dst + 28) = make_uint2(0u, 0u);   // zero pad cols 28..31
        }

        // ---- stage B: 64 filters x 4 kernel rows of weight (exactly 256 lines)
        // (f*P+p)*784 + kh*28 is a multiple of 4 floats -> 16B aligned: float4.
        {
            const int f   = tid >> 2;
            const int khl = tid & 3;
            const float* src = w + (size_t)(f * P_ + p) * K_
                                 + (size_t)(kh0 + khl) * KW_;
            _Float16* dst = &Bs[f * BROW + khl * 32];
            #pragma unroll
            for (int j = 0; j < 7; ++j) {
                float4 v = *(const float4*)(src + 4 * j);
                uint2 pkd; pkd.x = pkcvt(v.x, v.y); pkd.y = pkcvt(v.z, v.w);
                *(uint2*)(dst + 4 * j) = pkd;
            }
            *(uint2*)(dst + 28) = make_uint2(0u, 0u);
        }
        __syncthreads();

        // ---- compute: 4 K-steps of 32 (each = one zero-padded kernel row)
        #pragma unroll
        for (int kk = 0; kk < RPC; ++kk) {
            Frag a;
            const uint4* ar = (const uint4*)&As[(wave * 16 + nlo) * AROW];
            a.q[0] = ar[(kk * 32 + lbase) >> 3];        // K = k0 + {0|8} .. +7
            a.q[1] = ar[(kk * 32 + 16 + lbase) >> 3];   // K = k0+16 + {0|8} .. +7
            #pragma unroll
            for (int nt = 0; nt < 4; ++nt) {
                Frag bf;
                const uint4* br = (const uint4*)&Bs[(nt * 16 + nlo) * BROW];
                bf.q[0] = br[(kk * 32 + lbase) >> 3];
                bf.q[1] = br[(kk * 32 + 16 + lbase) >> 3];
                v8f& acc = nt == 0 ? acc0 : nt == 1 ? acc1 : nt == 2 ? acc2 : acc3;
                acc = __builtin_amdgcn_wmma_f32_16x16x32_f16(
                        /*neg_a=*/false, a.v, /*neg_b=*/false, bf.v,
                        /*c_mod=*/(short)0, acc, /*reuse_a=*/false, /*reuse_b=*/false);
            }
        }
        __syncthreads();
    }

    // ---- epilogue: D layout: VGPR r, lanes 0-15 -> M=r, lanes 16-31 -> M=r+8
    #pragma unroll
    for (int nt = 0; nt < 4; ++nt) {
        const v8f acc = nt == 0 ? acc0 : nt == 1 ? acc1 : nt == 2 ? acc2 : acc3;
        const int f = nt * 16 + nlo;
        const float bv = bias[(size_t)f * P_ + p];
        #pragma unroll
        for (int r = 0; r < 8; ++r) {
            const int b = wave * 16 + r + hi * 8;
            out[(size_t)b * NFP_ + (size_t)f * P_ + p] = acc[r] + bv;
        }
    }
}

extern "C" void kernel_launch(void* const* d_in, const int* in_sizes, int n_in,
                              void* d_out, int out_size, void* d_ws, size_t ws_size,
                              hipStream_t stream) {
    const float* x    = (const float*)d_in[0];
    const float* wgt  = (const float*)d_in[1];
    const float* bias = (const float*)d_in[2];
    float* out = (float*)d_out;
    lc_conv_wmma<<<dim3(P_), dim3(256), 0, stream>>>(x, wgt, bias, out);
}
